// DNL_74234214744693
// MI455X (gfx1250) — matrix operand order repaired and checked by reading, not compiled, not run on hardware
//
#include <hip/hip_runtime.h>
#include <math.h>

// Problem constants (reference: B=4, C=64, H=W=64 -> N=4096)
#define B_ 4
#define C_ 64
#define N_ 4096

typedef _Float16 half_t;
typedef __attribute__((ext_vector_type(16))) _Float16 v16h;
typedef __attribute__((ext_vector_type(8)))  float    v8f;

union HV16 { v16h v; uint4 q[2]; };

// ---- CDNA5 WMMA operand loaders (wave32, 16x16x32 f16) ----
// A-matrix 16x32 (rows striped in-lane): lane = row (lane&15); per lane the 16
// halves are K = k0 + 16*half .. +15, contiguous.
__device__ __forceinline__ v16h loadA16x32(const half_t* rowPtr, int k0, int halfId) {
    HV16 u;
    u.q[0] = *(const uint4*)(rowPtr + k0 + 16 * halfId);
    u.q[1] = *(const uint4*)(rowPtr + k0 + 16 * halfId + 8);
    return u.v;
}
// B-matrix 32x16 (cols striped across lanes): lane = col (lane&15); per lane the
// 16 halves are K = k0 + 16*half .. +15, contiguous.
__device__ __forceinline__ v16h loadB32x16(const half_t* colPtr, int k0, int halfId) {
    HV16 u;
    const half_t* p = colPtr + k0 + 16 * halfId;
    u.q[0] = *(const uint4*)(p);
    u.q[1] = *(const uint4*)(p + 8);
    return u.v;
}

#define WMMA_F32_F16(A, Bm, Cm) \
    __builtin_amdgcn_wmma_f32_16x16x32_f16(false, (A), false, (Bm), (short)0, (Cm), false, false)

// ================= Kernel A: per-(b,c) spatial mean of x =================
__global__ void DNL_xmean_kernel(const float* __restrict__ x, float* __restrict__ xmean) {
    __shared__ float red[256];
    const float* p = x + (size_t)blockIdx.x * N_;
    float s = 0.f;
    for (int i = threadIdx.x; i < N_; i += 256) s += p[i];
    red[threadIdx.x] = s;
    __syncthreads();
    for (int st = 128; st > 0; st >>= 1) {
        if (threadIdx.x < st) red[threadIdx.x] += red[threadIdx.x + st];
        __syncthreads();
    }
    if (threadIdx.x == 0) xmean[blockIdx.x] = red[0] * (1.0f / N_);
}

// ====== Kernel B: all five 1x1-conv projections (memory-bound GEMMs) ======
// k_whiten = kw @ (x - xmean)   -> kT  [B][N][C] f16  (A-operand layout)
// q_whiten = qw @ (x - xmean)   -> qT  [B][N][C] f16  (B-operand layout)
// v        = vw @ x + vb        -> vT  [B][C][N] f16  (A-operand layout)
// wx       = BN(ww @ x + wb)    -> wx  [B][C][N] f32
// m        = mw @ x + mb        -> mv  [B][N]    f32
__global__ void DNL_proj_kernel(const float* __restrict__ x,
    const float* __restrict__ qw, const float* __restrict__ kw,
    const float* __restrict__ mw, const float* __restrict__ mb1,
    const float* __restrict__ vw, const float* __restrict__ vb,
    const float* __restrict__ ww, const float* __restrict__ wb,
    const float* __restrict__ g, const float* __restrict__ beta,
    const float* __restrict__ rmean, const float* __restrict__ rvar,
    const float* __restrict__ xmean,
    half_t* __restrict__ kT, half_t* __restrict__ qT, half_t* __restrict__ vT,
    float* __restrict__ wx, float* __restrict__ mv)
{
    __shared__ float xs[64][65];
    __shared__ float xm[64];
    const int b  = blockIdx.x >> 6;
    const int n0 = (blockIdx.x & 63) << 6;
    const int t  = threadIdx.x;

    const float* xb = x + (size_t)b * C_ * N_;
    for (int i = t; i < 64 * 64; i += 256) {
        int ci = i >> 6, nn = i & 63;
        xs[ci][nn] = xb[(size_t)ci * N_ + n0 + nn];
    }
    if (t < 64) xm[t] = xmean[b * 64 + t];
    __syncthreads();

    const int co = t & 63;
    const int nq = t >> 6;  // 0..3, each thread owns 16 consecutive n
    float ak[16], aq[16], av[16], aw[16];
#pragma unroll
    for (int j = 0; j < 16; j++) { ak[j] = 0.f; aq[j] = 0.f; av[j] = 0.f; aw[j] = 0.f; }

    for (int ci = 0; ci < 64; ci++) {
        const float wk = kw[co * 64 + ci], wq = qw[co * 64 + ci];
        const float wv = vw[co * 64 + ci], w2 = ww[co * 64 + ci];
        const float xmv = xm[ci];
#pragma unroll
        for (int j = 0; j < 16; j++) {
            const float xr = xs[ci][nq * 16 + j];
            const float xc = xr - xmv;                 // centering == whitening (linear)
            ak[j] = fmaf(wk, xc, ak[j]);
            aq[j] = fmaf(wq, xc, aq[j]);
            av[j] = fmaf(wv, xr, av[j]);
            aw[j] = fmaf(w2, xr, aw[j]);
        }
    }
    const float vbias = vb[co], wbias = wb[co];
    const float inv   = g[co] * rsqrtf(rvar[co] + 1e-5f);
    const float shift = beta[co] - rmean[co] * inv;
#pragma unroll
    for (int j = 0; j < 16; j++) {
        const int n = n0 + nq * 16 + j;
        kT[((size_t)b * N_ + n) * 64 + co]   = (half_t)ak[j];
        qT[((size_t)b * N_ + n) * 64 + co]   = (half_t)aq[j];
        vT[((size_t)b * 64 + co) * N_ + n]   = (half_t)(av[j] + vbias);
        wx[((size_t)b * 64 + co) * N_ + n]   = (aw[j] + wbias) * inv + shift;
    }
    if (t < 64) {
        float acc = mb1[0];
        for (int ci = 0; ci < 64; ci++) acc = fmaf(mw[ci], xs[ci][t], acc);
        mv[(size_t)b * N_ + n0 + t] = acc;
    }
}

// ========= Kernel C (pass A): flash row stats of qk over last axis =========
// One block = 16 rows n; 8 waves sweep all 4096 columns m with 2 chained WMMAs
// per 16x16 tile. B tiles are register double-buffered (loads for tile i+1
// issue before the WMMAs of tile i), so the pre-WMMA loadcnt wait refers to
// loads launched a full iteration earlier. Lane-local online (max, sum-exp),
// single exp per element; cross-lane merge once at the end.
__global__ void DNL_rowstats_kernel(const half_t* __restrict__ kT,
                                    const half_t* __restrict__ qT,
                                    float2* __restrict__ rstats)
{
    __shared__ float2 st[8][16];
    const int b  = blockIdx.x >> 8;          // N/16 = 256 row-blocks per batch
    const int n0 = (blockIdx.x & 255) << 4;
    const int t = threadIdx.x, w = t >> 5, lane = t & 31;
    const int halfId = lane >> 4, lcol = lane & 15;

    const half_t* kbase = kT + (size_t)b * N_ * 64;
    const half_t* qbase = qT + (size_t)b * N_ * 64;

    const half_t* arow = kbase + (size_t)(n0 + lcol) * 64;
    const v16h a0 = loadA16x32(arow, 0, halfId);   // K = c 0..31
    const v16h a1 = loadA16x32(arow, 32, halfId);  // K = c 32..63

    float mL[8], sL[8];
#pragma unroll
    for (int r = 0; r < 8; r++) { mL[r] = -3.0e38f; sL[r] = 0.f; }

    const half_t* bcol0 = qbase + (size_t)(w * 16 + lcol) * 64;  // stride per it: 128*64
    v16h b0 = loadB32x16(bcol0, 0, halfId);
    v16h b1 = loadB32x16(bcol0, 32, halfId);

    for (int it = 0; it < 32; it++) {
        // issue next iteration's B tile before consuming this one
        const int itn = (it + 1 < 32) ? it + 1 : it;
        const half_t* bnx = bcol0 + (size_t)itn * 128 * 64;
        __builtin_prefetch(bcol0 + (size_t)((it + 2 < 32) ? it + 2 : itn) * 128 * 64, 0, 3);
        const v16h c0 = loadB32x16(bnx, 0, halfId);
        const v16h c1 = loadB32x16(bnx, 32, halfId);

        v8f d = {};
        d = WMMA_F32_F16(a0, b0, d);
        d = WMMA_F32_F16(a1, b1, d);
#pragma unroll
        for (int r = 0; r < 8; r++) {
            const float v   = d[r];
            const float dlt = v - mL[r];
            const float e   = __expf(-fabsf(dlt));       // one exp per element
            const bool  gt  = dlt > 0.f;
            sL[r] = gt ? fmaf(sL[r], e, 1.f) : (sL[r] + e);
            mL[r] = gt ? v : mL[r];
        }
        b0 = c0; b1 = c1;
    }

    // 16-lane (column-group) merge, once
#pragma unroll
    for (int r = 0; r < 8; r++) {
#pragma unroll
        for (int msk = 1; msk < 16; msk <<= 1) {
            const float om = __shfl_xor(mL[r], msk);
            const float os = __shfl_xor(sL[r], msk);
            const float nm = fmaxf(mL[r], om);
            sL[r] = sL[r] * __expf(mL[r] - nm) + os * __expf(om - nm);
            mL[r] = nm;
        }
    }
    if (lcol == 0) {
#pragma unroll
        for (int r = 0; r < 8; r++) st[w][halfId * 8 + r] = make_float2(mL[r], sL[r]);
    }
    __syncthreads();
    if (t < 16) {
        float cmax = -3.0e38f;
        for (int i = 0; i < 8; i++) cmax = fmaxf(cmax, st[i][t].x);
        float cs = 0.f;
        for (int i = 0; i < 8; i++) cs += st[i][t].y * __expf(st[i][t].x - cmax);
        rstats[(size_t)b * N_ + n0 + t] = make_float2(cmax, cs);
    }
}

// ===== Kernel D (pass B): y = v @ (softmax(qk) + softmax_batch(m m^T)), fused =====
// One block = batch b, 64 output columns m. Loop n in chunks of 32:
//   WMMA#1 recomputes qk tile, VALU applies exp/sum + rank-1 batch-softmax term,
//   f16 tile staged through double-buffered LDS in B-operand ([m][n]) layout,
//   WMMA#2 accumulates y. kT A-tiles register double-buffered. ONE barrier per
//   chunk: iter i stages chunk i+1 into LDS buf (i+1)%2, writes f[i%2], syncs,
//   then WMMA#2 reads f[i%2] — every cross-wave hazard spans exactly one barrier.
__global__ void DNL_attnout_kernel(const half_t* __restrict__ kT,
                                   const half_t* __restrict__ qT,
                                   const half_t* __restrict__ vT,
                                   const float* __restrict__ mv,
                                   const float2* __restrict__ rstats,
                                   const float* __restrict__ wx,
                                   float* __restrict__ out)
{
    __shared__ half_t fls[2][64][40];  // [buf][m_local][n_local], rows 16B-aligned
    __shared__ float2 rst[2][32];
    __shared__ float  mvn[2][4][32];
    __shared__ float  mvm[4][64];

    const int b  = blockIdx.x >> 6;
    const int m0 = (blockIdx.x & 63) << 6;
    const int t = threadIdx.x, w = t >> 5, lane = t & 31;
    const int halfId = lane >> 4, lcol = lane & 15;
    const int ns = w & 1, ms = w >> 1;   // WMMA#1 tile: n-sub 0..1, m-sub 0..3
    const int cs2 = w & 1;               // WMMA#2: pair of c-subtiles

    { // preload m-vector slice for this block's 64 columns (all 4 batches)
        const int bb = t >> 6, j = t & 63;
        mvm[bb][j] = mv[(size_t)bb * N_ + m0 + j];
    }
    // prologue: stage chunk 0 into buffer 0
    if (t < 32) rst[0][t] = rstats[(size_t)b * N_ + t];
    else if (t < 160) { const int i = t - 32; mvn[0][i >> 5][i & 31] = mv[(size_t)(i >> 5) * N_ + (i & 31)]; }

    const half_t* kbase = kT + (size_t)b * N_ * 64;
    const half_t* qbase = qT + (size_t)b * N_ * 64;
    const half_t* vbase = vT + (size_t)b * 64 * N_;

    // qT B-operand for this wave's m-subtile: invariant across the n loop
    const half_t* qcol = qbase + (size_t)(m0 + ms * 16 + lcol) * 64;
    const v16h bq0 = loadB32x16(qcol, 0, halfId);
    const v16h bq1 = loadB32x16(qcol, 32, halfId);

    v8f acc0 = {}, acc1 = {};
    const int ml = ms * 16 + lcol;
    const half_t* arow0 = kbase + (size_t)(ns * 16 + lcol) * 64;  // + nc*64 per chunk
    const half_t* vrow0 = vbase + (size_t)(cs2 * 32 + lcol) * N_;
    const half_t* vrow1 = vbase + (size_t)(cs2 * 32 + 16 + lcol) * N_;

    // register double-buffer for the kT A-tile
    v16h a0 = loadA16x32(arow0, 0, halfId);
    v16h a1 = loadA16x32(arow0, 32, halfId);
    __syncthreads();   // chunk-0 staging visible

    for (int it = 0; it < N_ / 32; it++) {
        const int nc  = it * 32;
        const int cur = it & 1, nxt = cur ^ 1;
        const int itn = (it + 1 < N_ / 32) ? it + 1 : it;

        // stage next chunk's row stats + m-vector slice into the other buffer
        if (t < 32) rst[nxt][t] = rstats[(size_t)b * N_ + itn * 32 + t];
        else if (t < 160) {
            const int i = t - 32;
            mvn[nxt][i >> 5][i & 31] = mv[(size_t)(i >> 5) * N_ + itn * 32 + (i & 31)];
        }
        // issue next chunk's kT A-tile; prefetch next V rows
        const half_t* anx = arow0 + (size_t)itn * 32 * 64;
        __builtin_prefetch(vrow0 + itn * 32, 0, 3);
        __builtin_prefetch(vrow1 + itn * 32, 0, 3);
        const v16h na0 = loadA16x32(anx, 0, halfId);
        const v16h na1 = loadA16x32(anx, 32, halfId);

        // WMMA#1: qk tile [16 n x 16 m], K = c = 64 (2 chained steps)
        v8f d = {};
        d = WMMA_F32_F16(a0, bq0, d);
        d = WMMA_F32_F16(a1, bq1, d);

        // softmax transform + cross-batch rank-1 softmax term -> f16 LDS [m][n]
#pragma unroll
        for (int r = 0; r < 8; r++) {
            const int nl = ns * 16 + r + 8 * halfId;       // D layout: M = r + 8*half
            const float2 rs = rst[cur][nl];
            const float p1 = __expf(d[r] - rs.x) / rs.y;
            const float s0 = mvn[cur][0][nl] * mvm[0][ml];
            const float s1 = mvn[cur][1][nl] * mvm[1][ml];
            const float s2 = mvn[cur][2][nl] * mvm[2][ml];
            const float s3 = mvn[cur][3][nl] * mvm[3][ml];
            const float mx = fmaxf(fmaxf(s0, s1), fmaxf(s2, s3));
            const float e0 = __expf(s0 - mx), e1 = __expf(s1 - mx);
            const float e2 = __expf(s2 - mx), e3 = __expf(s3 - mx);
            const float sum = e0 + e1 + e2 + e3;
            const float eb = (b == 0) ? e0 : (b == 1) ? e1 : (b == 2) ? e2 : e3;
            fls[cur][ml][nl] = (half_t)(p1 + eb / sum);
        }
        __syncthreads();   // single barrier per chunk

        // WMMA#2: B-operand from LDS ([m][n] => per-lane contiguous K), A = v tiles
        HV16 ub;
        const half_t* fp = &fls[cur][ml][16 * halfId];
        ub.q[0] = *(const uint4*)(fp);
        ub.q[1] = *(const uint4*)(fp + 8);
        const v16h bf = ub.v;

        const v16h aV0 = loadA16x32(vrow0, nc, halfId);
        acc0 = WMMA_F32_F16(aV0, bf, acc0);
        const v16h aV1 = loadA16x32(vrow1, nc, halfId);
        acc1 = WMMA_F32_F16(aV1, bf, acc1);

        a0 = na0; a1 = na1;
    }

    // epilogue: y + wx
    const float* wxb = wx + (size_t)b * 64 * N_;
    float* ob = out + (size_t)b * 64 * N_;
    const int mcol = m0 + ms * 16 + lcol;
#pragma unroll
    for (int r = 0; r < 8; r++) {
        const int c0 = cs2 * 32 + r + 8 * halfId;
        ob[(size_t)c0 * N_ + mcol] = acc0[r] + wxb[(size_t)c0 * N_ + mcol];
        const int c1 = cs2 * 32 + 16 + r + 8 * halfId;
        ob[(size_t)c1 * N_ + mcol] = acc1[r] + wxb[(size_t)c1 * N_ + mcol];
    }
}

extern "C" void kernel_launch(void* const* d_in, const int* in_sizes, int n_in,
                              void* d_out, int out_size, void* d_ws, size_t ws_size,
                              hipStream_t stream)
{
    (void)in_sizes; (void)n_in; (void)out_size; (void)ws_size;
    const float* x    = (const float*)d_in[0];
    const float* qw   = (const float*)d_in[1];
    // qb (d_in[2]) and kb (d_in[4]) cancel under whitening (k - mean(k))
    const float* kw   = (const float*)d_in[3];
    const float* mw   = (const float*)d_in[5];
    const float* mb1  = (const float*)d_in[6];
    const float* vw   = (const float*)d_in[7];
    const float* vb   = (const float*)d_in[8];
    const float* ww   = (const float*)d_in[9];
    const float* wb   = (const float*)d_in[10];
    const float* bng  = (const float*)d_in[11];
    const float* bnb  = (const float*)d_in[12];
    const float* bnrm = (const float*)d_in[13];
    const float* bnrv = (const float*)d_in[14];

    // workspace layout (~10.3 MB total)
    char* ws = (char*)d_ws;
    size_t o = 0;
    half_t* kT   = (half_t*)(ws + o); o += (size_t)B_ * N_ * C_ * 2;  // 2 MB
    half_t* qT   = (half_t*)(ws + o); o += (size_t)B_ * N_ * C_ * 2;  // 2 MB
    half_t* vT   = (half_t*)(ws + o); o += (size_t)B_ * N_ * C_ * 2;  // 2 MB
    float*  wx   = (float*)(ws + o);  o += (size_t)B_ * C_ * N_ * 4;  // 4 MB
    float*  mv   = (float*)(ws + o);  o += (size_t)B_ * N_ * 4;       // 64 KB
    float*  xmean= (float*)(ws + o);  o += (size_t)B_ * C_ * 4;       // 1 KB
    float2* rst  = (float2*)(ws + o); o += (size_t)B_ * N_ * 8;       // 128 KB

    DNL_xmean_kernel<<<B_ * C_, 256, 0, stream>>>(x, xmean);
    DNL_proj_kernel<<<B_ * (N_ / 64), 256, 0, stream>>>(
        x, qw, kw, mw, mb1, vw, vb, ww, wb, bng, bnb, bnrm, bnrv, xmean,
        kT, qT, vT, wx, mv);
    DNL_rowstats_kernel<<<B_ * (N_ / 16), 256, 0, stream>>>(kT, qT, rst);
    DNL_attnout_kernel<<<B_ * (N_ / 64), 256, 0, stream>>>(
        kT, qT, vT, mv, rst, wx, (float*)d_out);
}
